// PointerGeneratorNetwork_16621523435656
// MI455X (gfx1250) — matrix operand order repaired and checked by reading, not compiled, not run on hardware
//
#include <hip/hip_runtime.h>
#include <hip/hip_bf16.h>

// ---------------- problem constants ----------------
static constexpr int V    = 32000;
static constexpr int E    = 128;
static constexpr int H    = 256;
static constexpr int G4   = 4 * H;     // 1024 gate width
static constexpr int NENC = 2000;
static constexpr int TDEC = 64;
static constexpr int B    = 4;
static constexpr int U    = 50;
static constexpr int W    = 60;

typedef __attribute__((ext_vector_type(16))) __bf16 v16bf;
typedef __attribute__((ext_vector_type(8)))  float  v8f;

__device__ inline float sigf(float x) { return 1.0f / (1.0f + expf(-x)); }

// ---------------- CDNA5 async global->LDS copy (ASYNCcnt path) -------------
// One 16B chunk per lane, direct to LDS, no VGPR round trip.
__device__ inline void async_load_b128(const void* gsrc, void* ldst) {
  unsigned lds = (unsigned)(size_t)ldst;            // flat addr[31:0] == LDS offset
  unsigned long long g = (unsigned long long)(size_t)gsrc;
  asm volatile("global_load_async_to_lds_b128 %0, %1, off"
               :: "v"(lds), "v"(g) : "memory");
}
__device__ inline void wait_asynccnt_0() { asm volatile("s_wait_asynccnt 0x0" ::: "memory"); }
__device__ inline void wait_asynccnt_3() { asm volatile("s_wait_asynccnt 0x3" ::: "memory"); }

// ---------------- WMMA fragment helpers (CDNA5 16x16x32 bf16 layouts) --------
// A matrix 16x32 (MxK): lane = M row (mod 16); lanes 0-15 hold K {0..7,16..23},
// lanes 16-31 hold K {8..15,24..31}.
__device__ inline v16bf load_frag_a(const __bf16* a, int ld, int lane) {
  int m   = lane & 15;
  int klo = (lane < 16) ? 0  : 8;
  int khi = (lane < 16) ? 16 : 24;
  const __bf16* r = a + (size_t)m * ld;
  v16bf v;
#pragma unroll
  for (int e = 0; e < 8; ++e) v[e] = r[klo + e];
#pragma unroll
  for (int e = 0; e < 8; ++e) v[8 + e] = r[khi + e];
  return v;
}
// B matrix 32x16 (KxN) taken from row-major weight W[N][K]:
// lane = N col (mod 16); lanes 0-15 hold K {0..7,16..23}, lanes 16-31 hold K {8..15,24..31}.
__device__ inline v16bf load_frag_b(const __bf16* w, int ld, int lane) {
  int n  = lane & 15;
  int klo = (lane < 16) ? 0  : 8;
  int khi = (lane < 16) ? 16 : 24;
  const __bf16* r = w + (size_t)n * ld;
  v16bf v;
#pragma unroll
  for (int e = 0; e < 8; ++e) v[e] = r[klo + e];
#pragma unroll
  for (int e = 0; e < 8; ++e) v[8 + e] = r[khi + e];
  return v;
}

// ---------------- generic bf16 WMMA GEMM: C[M,N] = A[M,K] @ Wt[N,K]^T + bias -
// M % 64 == 0, N % 128 == 0, K % 32 == 0. Double-buffered async-to-LDS staging.
#define GB_M 64
#define GB_N 128
#define GB_K 32

__global__ __launch_bounds__(256) void gemm_bf16_kernel(
    const __bf16* __restrict__ A, const __bf16* __restrict__ Wt,
    const float* __restrict__ bias, float* __restrict__ C,
    int M, int N, int K)
{
  __shared__ __bf16 sA[2][GB_M * GB_K];
  __shared__ __bf16 sW[2][GB_N * GB_K];
  int tid  = threadIdx.x;
  int lane = tid & 31;
  int wave = tid >> 5;                 // 8 waves
  int m0 = blockIdx.y * GB_M;
  int n0 = blockIdx.x * GB_N;
  int wm = (wave & 3) * 16;            // wave's m-tile
  int wn = (wave >> 2) * 64;           // wave's 4-tile n-strip
  v8f acc[4] = {};

  // A tile 64x32: one 16B chunk per thread. W tile 128x32: two 16B chunks.
  int ar = tid >> 2, ac = (tid & 3) * 8;
  int wr = tid >> 1, wc = (tid & 1) * 16;

  auto stage = [&](int buf, int k0) {
    async_load_b128(A  + (size_t)(m0 + ar) * K + k0 + ac,     &sA[buf][ar * GB_K + ac]);
    async_load_b128(Wt + (size_t)(n0 + wr) * K + k0 + wc,     &sW[buf][wr * GB_K + wc]);
    async_load_b128(Wt + (size_t)(n0 + wr) * K + k0 + wc + 8, &sW[buf][wr * GB_K + wc + 8]);
  };

  stage(0, 0);
  int nk = K / GB_K;
  for (int ki = 0; ki < nk; ++ki) {
    int buf = ki & 1;
    bool hasNext = (ki + 1 < nk);
    if (hasNext) {
      stage(buf ^ 1, (ki + 1) * GB_K);   // prefetch next tile (stays in flight)
      wait_asynccnt_3();                 // current tile's 3 copies done (in-order)
    } else {
      wait_asynccnt_0();
    }
    __syncthreads();
    v16bf af = load_frag_a(&sA[buf][wm * GB_K], GB_K, lane);
#pragma unroll
    for (int i = 0; i < 4; ++i) {
      v16bf bf_ = load_frag_b(&sW[buf][(wn + i * 16) * GB_K], GB_K, lane);
      acc[i] = __builtin_amdgcn_wmma_f32_16x16x32_bf16(
          false, af, false, bf_, (short)0, acc[i], false, false);
    }
    __syncthreads();   // all waves done reading buf before it is restaged
  }
  // C/D layout: VGPR r -> M = r + (lane<16 ? 0 : 8); N = lane&15
  int nlane = lane & 15;
  int mb    = (lane < 16) ? 0 : 8;
#pragma unroll
  for (int i = 0; i < 4; ++i) {
    int nc = n0 + wn + i * 16 + nlane;
    float bv = bias ? bias[nc] : 0.0f;
#pragma unroll
    for (int r = 0; r < 8; ++r)
      C[(size_t)(m0 + wm + mb + r) * N + nc] = acc[i][r] + bv;
  }
}

// ---------------- persistent WMMA LSTM scan -----------------------------
// One block per direction (blockIdx.x: 0 => set0, 1 => set1), 512 threads =
// 16 waves. h kept in LDS as 16x256 bf16 A-matrix (rows B..15 stay zero).
// Per step: 16 waves x (8 k-steps x 4 n-tiles) WMMA -> gates (4x1024) in LDS,
// then fused sigmoid/tanh cell update. Whh streamed from L2 (bf16, 512KB).
__global__ __launch_bounds__(512) void lstm_scan_kernel(
    const float* xg0, const __bf16* Whh0, float* out0, int off0, int dir0,
    const float* xg1, const __bf16* Whh1, float* out1, int off1, int dir1,
    int steps, int outStride)
{
  __shared__ __bf16 hbf[16 * H];
  __shared__ float  cst[B * H];
  __shared__ float  gsh[B * G4];

  const float*  xg  = blockIdx.x ? xg1  : xg0;
  const __bf16* Whh = blockIdx.x ? Whh1 : Whh0;
  float*        out = blockIdx.x ? out1 : out0;
  int off = blockIdx.x ? off1 : off0;
  int dir = blockIdx.x ? dir1 : dir0;

  int tid = threadIdx.x, lane = tid & 31, wave = tid >> 5;   // 16 waves
  for (int i = tid; i < 16 * H; i += 512) hbf[i] = (__bf16)0.0f;
  for (int i = tid; i < B * H;  i += 512) cst[i] = 0.0f;
  __syncthreads();

  int n0 = wave * 64;
  for (int s = 0; s < steps; ++s) {
    int t = (dir > 0) ? s : (steps - 1 - s);
    v8f acc[4] = {};
#pragma unroll
    for (int k0 = 0; k0 < H; k0 += 32) {
      v16bf af = load_frag_a(hbf + k0, H, lane);
#pragma unroll
      for (int i = 0; i < 4; ++i) {
        v16bf bf_ = load_frag_b(Whh + (size_t)(n0 + i * 16) * H + k0, H, lane);
        acc[i] = __builtin_amdgcn_wmma_f32_16x16x32_bf16(
            false, af, false, bf_, (short)0, acc[i], false, false);
      }
    }
    if (lane < 16) {        // rows 0..3 = batch, VGPR r -> M=r for lanes 0-15
#pragma unroll
      for (int i = 0; i < 4; ++i) {
        int n = n0 + i * 16 + lane;
#pragma unroll
        for (int r = 0; r < B; ++r)
          gsh[r * G4 + n] = acc[i][r] + xg[((size_t)r * steps + t) * G4 + n];
      }
    }
    __syncthreads();
#pragma unroll
    for (int p = tid; p < B * H; p += 512) {
      int b = p >> 8, j = p & 255;
      float ig = gsh[b * G4 + j];
      float fg = gsh[b * G4 + H + j];
      float gg = gsh[b * G4 + 2 * H + j];
      float og = gsh[b * G4 + 3 * H + j];
      float c  = sigf(fg) * cst[b * H + j] + sigf(ig) * tanhf(gg);
      float h  = sigf(og) * tanhf(c);
      cst[b * H + j] = c;
      hbf[b * H + j] = (__bf16)h;
      out[((size_t)b * steps + t) * outStride + off + j] = h;
    }
    __syncthreads();
  }
}

// ---------------- pack input -> input_pgn, uw_len ------------------------
__global__ void pack_kernel(const int* __restrict__ input,
                            const int* __restrict__ u_len,
                            const int* __restrict__ w_len,
                            int* __restrict__ input_pgn, int* __restrict__ uw_len)
{
  int tid = threadIdx.x;
  for (int i = tid; i < B * NENC; i += blockDim.x) input_pgn[i] = 0;
  __syncthreads();
  if (tid < B) {
    int b = tid;
    int csum = 0, prefix = 0;
    int ul = u_len[b];
    for (int j = 0; j < U; ++j) {
      int wl = (j < ul) ? w_len[b * U + j] : 0;
      csum += wl;
      bool incl = (csum <= NENC) && (j < ul);
      int we = incl ? wl : 0;
      for (int w = 0; w < we; ++w)
        input_pgn[b * NENC + prefix + w] = input[(b * U + j) * W + w];
      prefix += we;
    }
    uw_len[b] = prefix;
  }
}

// ---------------- embedding gathers --------------------------------------
__global__ void embed_enc_kernel(const float* __restrict__ emb,
                                 const int* __restrict__ pgn,
                                 __bf16* __restrict__ Xbf, int total)
{
  int i = blockIdx.x * blockDim.x + threadIdx.x;
  if (i >= total) return;
  int tok = pgn[i >> 7];
  Xbf[i] = (__bf16)emb[(size_t)tok * E + (i & 127)];
}

__global__ void embed_dec_kernel(const float* __restrict__ emb,
                                 const int* __restrict__ target,
                                 __bf16* __restrict__ Xbf,
                                 float* __restrict__ Xf, int total)
{
  int i = blockIdx.x * blockDim.x + threadIdx.x;
  if (i >= total) return;
  int tok = target[i >> 7];
  float v = emb[(size_t)tok * E + (i & 127)];
  Xbf[i] = (__bf16)v;
  Xf[i]  = v;
}

// ---------------- casts ----------------------------------------------------
__global__ void cast_bf16_kernel(const float* __restrict__ in,
                                 __bf16* __restrict__ out, int n)
{
  int i = blockIdx.x * blockDim.x + threadIdx.x;
  if (i < n) out[i] = (__bf16)in[i];
}

__global__ void cast_rows_bf16_kernel(const float* __restrict__ in, int ldin,
                                      __bf16* __restrict__ out, int ldout,
                                      int dstOff, int rows, int cols)
{
  int i = blockIdx.x * blockDim.x + threadIdx.x;
  if (i >= rows * cols) return;
  int r = i / cols, c = i - r * cols;
  out[(size_t)r * ldout + dstOff + c] = (__bf16)in[(size_t)r * ldin + c];
}

// ---------------- attention + context + Pgen -------------------------------
// One block per (b,t). et[n]=v.tanh(WhHt[b,n,:]+WsSt[b,t,:]) masked softmax;
// context = at @ enc_out; Pgen = sigmoid(ctx.pcW + pcb + dec.psW + emb.pxW).
__global__ __launch_bounds__(256) void attention_kernel(
    const float* __restrict__ WhHt, const float* __restrict__ WsSt,
    const float* __restrict__ vvec, const float* __restrict__ enc_out,
    const int* __restrict__ uw_len, const float* __restrict__ dec_out,
    const float* __restrict__ dec_embed, const float* __restrict__ pcW,
    const float* __restrict__ pcb, const float* __restrict__ psW,
    const float* __restrict__ pxW, float* __restrict__ at,
    float* __restrict__ context, __bf16* __restrict__ cat_bf,
    float* __restrict__ Pgen)
{
  __shared__ float sWs[H], sv[H], red[256], sctx[2 * H];
  int bt = blockIdx.x;           // b*TDEC + t
  int b  = bt >> 6;
  int tid = threadIdx.x;
  sWs[tid] = WsSt[(size_t)bt * H + tid];
  sv[tid]  = vvec[tid];
  __syncthreads();

  int L = uw_len[b];
  float* atrow = at + (size_t)bt * NENC;
  float lmax = -__builtin_inff();
  for (int n = tid; n < NENC; n += 256) {
    float e;
    if (n < L) {
      const float* wh = WhHt + ((size_t)b * NENC + n) * H;
      float s = 0.f;
      for (int h = 0; h < H; ++h) s += sv[h] * tanhf(wh[h] + sWs[h]);
      e = s;
    } else {
      e = -__builtin_inff();
    }
    atrow[n] = e;
    lmax = fmaxf(lmax, e);
  }
  red[tid] = lmax; __syncthreads();
  for (int s = 128; s > 0; s >>= 1) {
    if (tid < s) red[tid] = fmaxf(red[tid], red[tid + s]);
    __syncthreads();
  }
  float gmax = red[0]; __syncthreads();

  float lsum = 0.f;
  for (int n = tid; n < NENC; n += 256) {
    float e = atrow[n];
    float w = (e == -__builtin_inff()) ? 0.0f : expf(e - gmax);
    atrow[n] = w; lsum += w;
  }
  red[tid] = lsum; __syncthreads();
  for (int s = 128; s > 0; s >>= 1) {
    if (tid < s) red[tid] += red[tid + s];
    __syncthreads();
  }
  float inv = 1.0f / red[0]; __syncthreads();
  for (int n = tid; n < NENC; n += 256) atrow[n] *= inv;
  __threadfence_block();
  __syncthreads();

  // context over 512 dims
  for (int h = tid; h < 2 * H; h += 256) {
    const float* encb = enc_out + (size_t)b * NENC * (2 * H) + h;
    float s = 0.f;
    for (int n = 0; n < NENC; ++n) s += atrow[n] * encb[(size_t)n * (2 * H)];
    sctx[h] = s;
    context[(size_t)bt * (2 * H) + h] = s;
    cat_bf[(size_t)bt * (3 * H) + H + h] = (__bf16)s;
  }
  __syncthreads();

  float pp = sctx[tid] * pcW[tid] + sctx[tid + H] * pcW[tid + H];
  pp += dec_out[(size_t)bt * H + tid] * psW[tid];
  if (tid < E) pp += dec_embed[(size_t)bt * E + tid] * pxW[tid];
  red[tid] = pp; __syncthreads();
  for (int s = 128; s > 0; s >>= 1) {
    if (tid < s) red[tid] += red[tid + s];
    __syncthreads();
  }
  if (tid == 0) Pgen[bt] = sigf(red[0] + pcb[0]);
}

// ---------------- final: pointer mix + log-softmax over V ------------------
__global__ __launch_bounds__(256) void final_kernel(
    const float* __restrict__ logits, const float* __restrict__ at,
    const float* __restrict__ Pgen, const int* __restrict__ target,
    const int* __restrict__ input_pgn, float* __restrict__ out)
{
  __shared__ float red[256];
  int r = blockIdx.x;            // b*TDEC + t
  int b = r >> 6, t = r & 63;
  int tid = threadIdx.x;

  int y = (t < TDEC - 1) ? target[b * TDEC + t + 1] : -1;
  const float* atrow = at + (size_t)r * NENC;
  const int*   pg    = input_pgn + b * NENC;
  float lv = 0.f;
  for (int n = tid; n < NENC; n += 256)
    if (pg[n] == y) lv += atrow[n];
  red[tid] = lv; __syncthreads();
  for (int s = 128; s > 0; s >>= 1) {
    if (tid < s) red[tid] += red[tid + s];
    __syncthreads();
  }
  float val = red[0]; __syncthreads();

  const float* lrow = logits + (size_t)r * V;
  float lmax = -__builtin_inff();
  for (int v_ = tid; v_ < V; v_ += 256) lmax = fmaxf(lmax, lrow[v_]);
  red[tid] = lmax; __syncthreads();
  for (int s = 128; s > 0; s >>= 1) {
    if (tid < s) red[tid] = fmaxf(red[tid], red[tid + s]);
    __syncthreads();
  }
  float gm = red[0]; __syncthreads();

  float lsum = 0.f;
  for (int v_ = tid; v_ < V; v_ += 256) lsum += expf(lrow[v_] - gm);
  red[tid] = lsum; __syncthreads();
  for (int s = 128; s > 0; s >>= 1) {
    if (tid < s) red[tid] += red[tid + s];
    __syncthreads();
  }
  float inv = 1.0f / red[0];

  float pgen = Pgen[r];
  float* orow = out + (size_t)r * V;
  for (int v_ = tid; v_ < V; v_ += 256) {
    float pv = expf(lrow[v_] - gm) * inv;
    float pw = pgen * pv + ((v_ == y) ? (1.0f - pgen) * val : 0.0f);
    orow[v_] = logf(pw);
  }
}

// ------------------------------------------------------------------------
extern "C" void kernel_launch(void* const* d_in, const int* in_sizes, int n_in,
                              void* d_out, int out_size, void* d_ws, size_t ws_size,
                              hipStream_t stream) {
  (void)in_sizes; (void)n_in; (void)out_size; (void)ws_size;
  // inputs (setup_inputs dict order, params flattened in _init_params order)
  const int*   input    = (const int*)d_in[0];
  const int*   u_len    = (const int*)d_in[1];
  const int*   w_len    = (const int*)d_in[2];
  const int*   target   = (const int*)d_in[3];
  const float* emb_enc  = (const float*)d_in[4];
  const float* emb_dec  = (const float*)d_in[5];
  const float* Wih_f    = (const float*)d_in[6];
  const float* Whh_f    = (const float*)d_in[7];
  const float* b_f      = (const float*)d_in[8];
  const float* Wih_b    = (const float*)d_in[9];
  const float* Whh_b    = (const float*)d_in[10];
  const float* b_b      = (const float*)d_in[11];
  const float* dWih     = (const float*)d_in[12];
  const float* dWhh     = (const float*)d_in[13];
  const float* d_b      = (const float*)d_in[14];
  const float* Wh       = (const float*)d_in[15];
  const float* Ws       = (const float*)d_in[16];
  const float* bs       = (const float*)d_in[17];
  const float* vvec     = (const float*)d_in[18];
  const float* L1W      = (const float*)d_in[19];
  const float* L1b      = (const float*)d_in[20];
  const float* L2W      = (const float*)d_in[21];
  const float* L2b      = (const float*)d_in[22];
  const float* pcW      = (const float*)d_in[23];
  const float* pcb      = (const float*)d_in[24];
  const float* psW      = (const float*)d_in[25];
  const float* pxW      = (const float*)d_in[26];
  float* out = (float*)d_out;

  char* p = (char*)d_ws;
  auto alloc = [&](size_t bytes) -> void* {
    void* r = (void*)p;
    p += (bytes + 255) & ~(size_t)255;
    return r;
  };
  const size_t BN = (size_t)B * NENC;   // 8000
  const size_t BT = (size_t)B * TDEC;   // 256

  int*    input_pgn = (int*)   alloc(BN * sizeof(int));
  int*    uw_len    = (int*)   alloc(B * sizeof(int));
  __bf16* Xenc_bf   = (__bf16*)alloc(BN * E * 2);
  __bf16* Xdec_bf   = (__bf16*)alloc(BT * E * 2);
  float*  dec_emb   = (float*) alloc(BT * E * 4);
  float*  xgf       = (float*) alloc(BN * G4 * 4);
  float*  xgb       = (float*) alloc(BN * G4 * 4);
  float*  xgd       = (float*) alloc(BT * G4 * 4);
  float*  enc_out   = (float*) alloc(BN * 2 * H * 4);
  __bf16* enc_bf    = (__bf16*)alloc(BN * 2 * H * 2);
  float*  dec_out   = (float*) alloc(BT * H * 4);
  __bf16* dec_bf    = (__bf16*)alloc(BT * H * 2);
  float*  WhHt      = (float*) alloc(BN * H * 4);
  float*  WsSt      = (float*) alloc(BT * H * 4);
  float*  at        = (float*) alloc(BT * NENC * 4);
  float*  context   = (float*) alloc(BT * 2 * H * 4);
  __bf16* cat_bf    = (__bf16*)alloc(BT * 3 * H * 2);
  float*  hid       = (float*) alloc(BT * H * 4);
  __bf16* hid_bf    = (__bf16*)alloc(BT * H * 2);
  float*  logits    = (float*) alloc(BT * V * 4);
  float*  Pgen      = (float*) alloc(BT * 4);
  __bf16* Wih_f_bf  = (__bf16*)alloc((size_t)G4 * E * 2);
  __bf16* Wih_b_bf  = (__bf16*)alloc((size_t)G4 * E * 2);
  __bf16* Whh_f_bf  = (__bf16*)alloc((size_t)G4 * H * 2);
  __bf16* Whh_b_bf  = (__bf16*)alloc((size_t)G4 * H * 2);
  __bf16* dWih_bf   = (__bf16*)alloc((size_t)G4 * E * 2);
  __bf16* dWhh_bf   = (__bf16*)alloc((size_t)G4 * H * 2);
  __bf16* Wh_bf     = (__bf16*)alloc((size_t)H * 2 * H * 2);
  __bf16* Ws_bf     = (__bf16*)alloc((size_t)H * H * 2);
  __bf16* L1_bf     = (__bf16*)alloc((size_t)H * 3 * H * 2);
  __bf16* L2_bf     = (__bf16*)alloc((size_t)V * H * 2);

  auto cast = [&](const float* src, __bf16* dst, int n) {
    cast_bf16_kernel<<<(n + 255) / 256, 256, 0, stream>>>(src, dst, n);
  };
  // weight conversions to bf16
  cast(Wih_f, Wih_f_bf, G4 * E);
  cast(Wih_b, Wih_b_bf, G4 * E);
  cast(Whh_f, Whh_f_bf, G4 * H);
  cast(Whh_b, Whh_b_bf, G4 * H);
  cast(dWih,  dWih_bf,  G4 * E);
  cast(dWhh,  dWhh_bf,  G4 * H);
  cast(Wh,    Wh_bf,    H * 2 * H);
  cast(Ws,    Ws_bf,    H * H);
  cast(L1W,   L1_bf,    H * 3 * H);
  cast(L2W,   L2_bf,    V * H);

  // pack + embeddings
  pack_kernel<<<1, 256, 0, stream>>>(input, u_len, w_len, input_pgn, uw_len);
  {
    int tot = (int)(BN * E);
    embed_enc_kernel<<<(tot + 255) / 256, 256, 0, stream>>>(emb_enc, input_pgn, Xenc_bf, tot);
  }
  {
    int tot = (int)(BT * E);
    embed_dec_kernel<<<(tot + 255) / 256, 256, 0, stream>>>(emb_dec, target, Xdec_bf, dec_emb, tot);
  }

  // input-projection GEMMs: xg = X @ Wih^T + b
  gemm_bf16_kernel<<<dim3(G4 / GB_N, (int)(BN / GB_M)), 256, 0, stream>>>(
      Xenc_bf, Wih_f_bf, b_f, xgf, (int)BN, G4, E);
  gemm_bf16_kernel<<<dim3(G4 / GB_N, (int)(BN / GB_M)), 256, 0, stream>>>(
      Xenc_bf, Wih_b_bf, b_b, xgb, (int)BN, G4, E);
  gemm_bf16_kernel<<<dim3(G4 / GB_N, (int)(BT / GB_M)), 256, 0, stream>>>(
      Xdec_bf, dWih_bf, d_b, xgd, (int)BT, G4, E);

  // bidirectional encoder scan (fwd block 0, bwd block 1, concurrent)
  lstm_scan_kernel<<<2, 512, 0, stream>>>(
      xgf, Whh_f_bf, enc_out, 0, +1,
      xgb, Whh_b_bf, enc_out, H, -1,
      NENC, 2 * H);
  // decoder scan
  lstm_scan_kernel<<<1, 512, 0, stream>>>(
      xgd, dWhh_bf, dec_out, 0, +1,
      xgd, dWhh_bf, dec_out, 0, +1,
      TDEC, H);

  // activation casts
  cast(enc_out, enc_bf, (int)(BN * 2 * H));
  cast(dec_out, dec_bf, (int)(BT * H));
  {
    int tot = (int)(BT * H);
    cast_rows_bf16_kernel<<<(tot + 255) / 256, 256, 0, stream>>>(
        dec_out, H, cat_bf, 3 * H, 0, (int)BT, H);
  }

  // attention projections
  gemm_bf16_kernel<<<dim3(2 * H / GB_N, (int)(BN / GB_M)), 256, 0, stream>>>(
      enc_bf, Wh_bf, nullptr, WhHt, (int)BN, H, 2 * H);
  gemm_bf16_kernel<<<dim3(2 * H / GB_N, (int)(BT / GB_M)), 256, 0, stream>>>(
      dec_bf, Ws_bf, bs, WsSt, (int)BT, H, H);

  // attention + context + Pgen
  attention_kernel<<<(int)BT, 256, 0, stream>>>(
      WhHt, WsSt, vvec, enc_out, uw_len, dec_out, dec_emb,
      pcW, pcb, psW, pxW, at, context, cat_bf, Pgen);

  // hid = cat @ L1^T + b ; logits = hid @ L2^T + b
  gemm_bf16_kernel<<<dim3(2 * H / GB_N, (int)(BT / GB_M)), 256, 0, stream>>>(
      cat_bf, L1_bf, L1b, hid, (int)BT, H, 3 * H);
  cast(hid, hid_bf, (int)(BT * H));
  gemm_bf16_kernel<<<dim3(V / GB_N, (int)(BT / GB_M)), 256, 0, stream>>>(
      hid_bf, L2_bf, L2b, logits, (int)BT, V, H);

  // pointer-generator mix + log
  final_kernel<<<(int)BT, 256, 0, stream>>>(logits, at, Pgen, target, input_pgn, out);
}
